// GraphAttentionLayer_83846351552830
// MI455X (gfx1250) — matrix operand order repaired and checked by reading, not compiled, not run on hardware
//
#include <hip/hip_runtime.h>
#include <hip/hip_bf16.h>

// ---------------------------------------------------------------------------
// GAT layer for MI455X (gfx1250, wave32, WMMA).
// B=4, N=4096, Fin=Fout=64.  Dominant op: attention(B,N,N) @ Wh(B,N,64)
// done as f16 WMMA (v_wmma_f32_16x16x32_f16) with f32 accumulation.
// ---------------------------------------------------------------------------

typedef __attribute__((ext_vector_type(16))) _Float16 v16h;
typedef __attribute__((ext_vector_type(8)))  _Float16 h8;
typedef __attribute__((ext_vector_type(4)))  _Float16 h4;
typedef __attribute__((ext_vector_type(8)))  float    v8f;

constexpr int Bn  = 4;
constexpr int Nn  = 4096;
constexpr int Fn  = 64;
constexpr int PSTR  = 1032;   // 1024 + 8 halves padding -> conflict-free ds_b128
constexpr int CHUNK = 1024;   // m-chunk held in LDS per iteration

#define ALPHA_LR 0.2f
#define NEGBIG  (-1.0e9f)

__device__ __forceinline__ float gat_e(float s1v, float s2v, int av) {
    float x = s1v + s2v;
    x = x > 0.f ? x : ALPHA_LR * x;       // leaky_relu
    return av ? x : x + NEGBIG;           // adjacency mask
}

// ---------------------------------------------------------------------------
// Kernel 1: Wh = h @ W  -> store transposed f16  WhT[b][f][m]
//           s1[b][m] = Wh . a1 ,  s2[b][m] = Wh . a2   (f32)
// 64 threads/block, each thread owns one row m.  0.13 GFLOP total; trivial.
// ---------------------------------------------------------------------------
__global__ void gat_wh_kernel(const float* __restrict__ h,
                              const float* __restrict__ W,
                              const float* __restrict__ a,
                              _Float16*    __restrict__ WhT,
                              float*       __restrict__ s1,
                              float*       __restrict__ s2)
{
    __shared__ float Ws[64 * 64];
    __shared__ float a1s[64], a2s[64];

    const int t = threadIdx.x;            // 0..63
    const int b = blockIdx.y;
    const int m = blockIdx.x * 64 + t;

    for (int i = 0; i < 64; ++i) Ws[i * 64 + t] = W[i * 64 + t];
    a1s[t] = a[t];
    a2s[t] = a[64 + t];
    __syncthreads();

    float hrow[64];
    const float4* hp = (const float4*)(h + ((size_t)(b * Nn + m)) * 64);
#pragma unroll
    for (int i = 0; i < 16; ++i) {
        float4 v = hp[i];
        hrow[4 * i + 0] = v.x; hrow[4 * i + 1] = v.y;
        hrow[4 * i + 2] = v.z; hrow[4 * i + 3] = v.w;
    }

    float s1v = 0.f, s2v = 0.f;
    for (int f = 0; f < 64; ++f) {
        float acc = 0.f;
#pragma unroll
        for (int i = 0; i < 64; ++i) acc += hrow[i] * Ws[i * 64 + f];
        WhT[(((size_t)(b * 64 + f)) << 12) + m] = (_Float16)acc;
        s1v += acc * a1s[f];
        s2v += acc * a2s[f];
    }
    s1[b * Nn + m] = s1v;
    s2[b * Nn + m] = s2v;
}

// ---------------------------------------------------------------------------
// Kernel 2: fused masked-softmax + (attention @ Wh) + ELU.
// One block (128 threads = 4 waves) per (b, 16-row tile).
//   Pass 1: per-row max & sum of exp(e) over all 4096 m (online, 8 thr/row).
//   Pass 2: 4 chunks of 1024 m: write P=softmax probs (f16) into LDS tile,
//           then each wave does 32 x v_wmma_f32_16x16x32_f16 for its f-tile.
//   Epilogue: ELU from the C/D fragment layout.
// ---------------------------------------------------------------------------
__global__ void gat_attn_kernel(const int*      __restrict__ adj,
                                const _Float16* __restrict__ WhT,
                                const float*    __restrict__ s1,
                                const float*    __restrict__ s2,
                                float*          __restrict__ out)
{
    __shared__ _Float16 Plds[16 * PSTR];      // 33 KB probability tile
    __shared__ float redMx[16][8];
    __shared__ float redSm[16][8];
    __shared__ float rowMax[16];
    __shared__ float rowInv[16];

    const int t   = threadIdx.x;              // 0..127
    const int b   = blockIdx.y;
    const int n0  = blockIdx.x * 16;
    const int r   = t >> 3;                   // row within tile (16 rows)
    const int sub = t & 7;                    // 8 threads per row

    const float s1v = s1[b * Nn + n0 + r];
    const int*   adjRow = adj + (size_t)(n0 + r) * Nn;
    const float* s2b    = s2 + b * Nn;

    // ---------------- Pass 1: online row max / sum(exp) -------------------
    float mx = -3.0e38f, sm = 0.f;
    const int m0 = sub * 512;
    for (int m = m0; m < m0 + 512; m += 4) {
        int4   av = *(const int4*)(adjRow + m);
        float4 sv = *(const float4*)(s2b + m);
        float xs[4] = { gat_e(s1v, sv.x, av.x), gat_e(s1v, sv.y, av.y),
                        gat_e(s1v, sv.z, av.z), gat_e(s1v, sv.w, av.w) };
#pragma unroll
        for (int j = 0; j < 4; ++j) {
            float x = xs[j];
            if (x > mx) { sm = sm * __expf(mx - x) + 1.f; mx = x; }
            else        { sm += __expf(x - mx); }
        }
    }
    redMx[r][sub] = mx;
    redSm[r][sub] = sm;
    __syncthreads();
    if (t < 16) {
        float M = redMx[t][0], S = redSm[t][0];
#pragma unroll
        for (int j = 1; j < 8; ++j) {
            float mj = redMx[t][j], sj = redSm[t][j];
            if (mj > M) { S = S * __expf(M - mj) + sj; M = mj; }
            else        { S += sj * __expf(mj - M); }
        }
        rowMax[t] = M;
        rowInv[t] = 1.f / S;
    }
    __syncthreads();

    // ---------------- per-wave WMMA setup ----------------------------------
    const int lane = t & 31;
    const int w    = t >> 5;                  // 4 waves, one 16-col f-tile each
    const int fcol = (w << 4) + (lane & 15);
    const _Float16* whB = WhT + (((size_t)(b * 64 + fcol)) << 12);
    const int arow = lane & 15;
    const int kbA  = (lane < 16) ? 0 : 16;    // 16-bit A frag K-run base
    const int kbB  = (lane < 16) ? 0 : 16;    // 16-bit B frag K base

    v8f acc = {};

    union AU { v16h v; h8 h[2]; };

    const float Mr = rowMax[r];
    const float Ir = rowInv[r];

    // ---------------- Pass 2: chunked P-fill + WMMA ------------------------
    for (int c = 0; c < 4; ++c) {
        const int mb = c * CHUNK;

        // fill P[16][CHUNK] in LDS (f16, normalized probabilities)
        const int mloc0 = sub * (CHUNK / 8);
        for (int mm = mloc0; mm < mloc0 + (CHUNK / 8); mm += 4) {
            const int m = mb + mm;
            int4   av = *(const int4*)(adjRow + m);
            float4 sv = *(const float4*)(s2b + m);
            h4 pv;
            pv[0] = (_Float16)(__expf(gat_e(s1v, sv.x, av.x) - Mr) * Ir);
            pv[1] = (_Float16)(__expf(gat_e(s1v, sv.y, av.y) - Mr) * Ir);
            pv[2] = (_Float16)(__expf(gat_e(s1v, sv.z, av.z) - Mr) * Ir);
            pv[3] = (_Float16)(__expf(gat_e(s1v, sv.w, av.w) - Mr) * Ir);
            *(h4*)&Plds[r * PSTR + mm] = pv;
        }
        __syncthreads();

        // WMMA: D(16x16) += P(16x32) @ Wh(32x16), 32 K-steps per chunk
        for (int k0 = 0; k0 < CHUNK; k0 += 32) {
            AU au;
            au.h[0] = *(const h8*)&Plds[arow * PSTR + k0 + kbA];       // K 0..7 / 16..23
            au.h[1] = *(const h8*)&Plds[arow * PSTR + k0 + kbA + 8];   // K 8..15 / 24..31
            v16h bf = *(const v16h*)(whB + mb + k0 + kbB);             // 16 consecutive K
            acc = __builtin_amdgcn_wmma_f32_16x16x32_f16(
                      false, au.v, false, bf, (short)0, acc, false, false);
        }
        __syncthreads();
    }

    // ---------------- Epilogue: ELU + store --------------------------------
#pragma unroll
    for (int rv = 0; rv < 8; ++rv) {
        const int row = rv + ((lane < 16) ? 0 : 8);
        float x = acc[rv];
        float y = x > 0.f ? x : (__expf(x) - 1.f);
        out[(((size_t)(b * Nn + n0 + row)) << 6) + fcol] = y;
    }
}

// ---------------------------------------------------------------------------
extern "C" void kernel_launch(void* const* d_in, const int* in_sizes, int n_in,
                              void* d_out, int out_size, void* d_ws, size_t ws_size,
                              hipStream_t stream)
{
    const float* h   = (const float*)d_in[0];   // (4, 4096, 64)
    const float* W   = (const float*)d_in[1];   // (64, 64)
    const float* a   = (const float*)d_in[2];   // (128, 1)
    const int*   adj = (const int*)d_in[3];     // (4096, 4096)
    float*       out = (float*)d_out;           // (4, 4096, 64)

    char* ws = (char*)d_ws;
    _Float16* WhT = (_Float16*)ws;                                  // B*F*N f16 = 2 MB
    float* s1 = (float*)(ws + (size_t)Bn * Fn * Nn * sizeof(_Float16));
    float* s2 = s1 + Bn * Nn;

    gat_wh_kernel<<<dim3(Nn / 64, Bn), 64, 0, stream>>>(h, W, a, WhT, s1, s2);
    gat_attn_kernel<<<dim3(Nn / 16, Bn), 128, 0, stream>>>(adj, WhT, s1, s2, out);
}
